// DeepSeekMoE_60026462929320
// MI455X (gfx1250) — matrix operand order repaired and checked by reading, not compiled, not run on hardware
//
#include <hip/hip_runtime.h>
#include <hip/hip_bf16.h>

// ---- constants (from reference) ----
#define SEQ    2048
#define BATCH  4
#define HIDDEN 2048
#define FFN    1408
#define FFN2   2816      // 2*FFN
#define NTOK   8192      // BATCH*SEQ
#define NEXP   8

typedef __attribute__((ext_vector_type(16))) __bf16 v16bf;
typedef __attribute__((ext_vector_type(8)))  float  v8f;

union FragU { v16bf v; uint4 q[2]; };

#define LDST 20   // LDS row stride in dwords (16 data + 4 pad, keeps 16B alignment)

__device__ __forceinline__ unsigned short f2bf(float f) {
    unsigned u = __float_as_uint(f);
    u += 0x7FFFu + ((u >> 16) & 1u);           // round-to-nearest-even
    return (unsigned short)(u >> 16);
}
__device__ __forceinline__ unsigned pack_bf16(float a, float b) {
    return (unsigned)f2bf(a) | ((unsigned)f2bf(b) << 16);
}

// async 16-byte copy global -> LDS (ASYNCcnt-tracked, CDNA5)
__device__ __forceinline__ void async_copy16(unsigned lds_off, const void* g) {
    unsigned long long ga = (unsigned long long)(uintptr_t)g;
    asm volatile("global_load_async_to_lds_b128 %0, %1, off"
                 :: "v"(lds_off), "v"(ga) : "memory");
}
__device__ __forceinline__ void wait_async0() {
    asm volatile("s_wait_asynccnt 0x0" ::: "memory");
}

// ---------------- zero output ----------------
__global__ __launch_bounds__(256) void moe_zero_kernel(float4* __restrict__ out) {
    out[(size_t)blockIdx.x * 256 + threadIdx.x] = make_float4(0.f, 0.f, 0.f, 0.f);
}

// ---------------- fp32 -> bf16 converters ----------------
// x[S,B,H] -> xb[t][h] (t = b*SEQ + s), one block per token row
__global__ __launch_bounds__(256) void moe_convx_kernel(const float* __restrict__ x,
                                                        unsigned* __restrict__ xb) {
    const int t = blockIdx.x;
    const int bb = t >> 11, ss = t & (SEQ - 1);
    const float* src = x + ((size_t)ss * BATCH + bb) * HIDDEN;
    unsigned* dst = xb + (size_t)t * (HIDDEN / 2);
#pragma unroll
    for (int v = 0; v < 4; ++v) {
        int j = threadIdx.x + 256 * v;          // 0..1023
        dst[j] = pack_bf16(src[2 * j], src[2 * j + 1]);
    }
}

// plain fp32 -> packed bf16 (same layout), 1024 dwords per block
__global__ __launch_bounds__(256) void moe_convw_kernel(const float* __restrict__ src,
                                                        unsigned* __restrict__ dst) {
#pragma unroll
    for (int v = 0; v < 4; ++v) {
        size_t j = (size_t)blockIdx.x * 1024 + threadIdx.x + 256 * v;
        dst[j] = pack_bf16(src[2 * j], src[2 * j + 1]);
    }
}

// ---------------- gate: softmax + top-2 combine weights ----------------
__global__ __launch_bounds__(256) void moe_gate_kernel(const float* __restrict__ x,
                                                       const float* __restrict__ wg,
                                                       float* __restrict__ C) {
    const int lane = threadIdx.x & 31;
    const int wave = threadIdx.x >> 5;
    const int t = blockIdx.x * 8 + wave;
    const int bb = t >> 11;
    const int ss = t & (SEQ - 1);
    const float* xr = x + ((size_t)ss * BATCH + bb) * HIDDEN;

    float acc[NEXP];
#pragma unroll
    for (int e = 0; e < NEXP; ++e) acc[e] = 0.f;

    for (int h = lane; h < HIDDEN; h += 32) {
        float xv = xr[h];
#pragma unroll
        for (int e = 0; e < NEXP; ++e)
            acc[e] = fmaf(xv, wg[e * HIDDEN + h], acc[e]);
    }
#pragma unroll
    for (int e = 0; e < NEXP; ++e)
#pragma unroll
        for (int off = 16; off > 0; off >>= 1)
            acc[e] += __shfl_xor(acc[e], off, 32);

    float m = acc[0];
#pragma unroll
    for (int e = 1; e < NEXP; ++e) m = fmaxf(m, acc[e]);
    float p[NEXP]; float sum = 0.f;
#pragma unroll
    for (int e = 0; e < NEXP; ++e) { p[e] = __expf(acc[e] - m); sum += p[e]; }
    float inv = __builtin_amdgcn_rcpf(sum);
#pragma unroll
    for (int e = 0; e < NEXP; ++e) p[e] *= inv;

    int i1 = 0;
#pragma unroll
    for (int e = 1; e < NEXP; ++e) if (p[e] > p[i1]) i1 = e;
    int i2 = (i1 == 0) ? 1 : 0;
#pragma unroll
    for (int e = 0; e < NEXP; ++e) if (e != i1 && p[e] > p[i2]) i2 = e;
    float norm = __builtin_amdgcn_rcpf(p[i1] + p[i2] + 1e-20f);

    if (lane < NEXP) {
        float cw = (lane == i1) ? p[i1] * norm : ((lane == i2) ? p[i2] * norm : 0.f);
        C[(size_t)t * NEXP + lane] = cw;
    }
}

// ---------------- GEMM1 + swiglu: act = silu(x@w1a) * (x@w1b), bf16 out ----------------
// block: 128 tokens x 64 F-cols (both halves); wave: 64 x 16 with a/b frags in registers
// double-buffered LDS; A tiles via global_load_async_to_lds_b128
__global__ __launch_bounds__(256) void moe_gemm1_kernel(const unsigned* __restrict__ xb,
                                                        const unsigned short* __restrict__ w1b,
                                                        unsigned short* __restrict__ act) {
    __shared__ __attribute__((aligned(16))) unsigned ldsA[2][128 * LDST];
    __shared__ __attribute__((aligned(16))) unsigned ldsB[2][128 * LDST];

    const int tid  = threadIdx.x;
    const int lane = tid & 31;
    const int wave = tid >> 5;
    const int wm   = wave >> 2;     // 0..1
    const int wn   = wave & 3;      // 0..3
    const int m0   = blockIdx.y * 128;
    const int n0   = blockIdx.x * 64;

    // B-staging assignment: thread -> (kpB 0..15, ngB 0..15); 8 cols each
    const int kpB = tid >> 4;
    const int ngB = tid & 15;
    const int colB = (ngB < 8) ? (n0 + ngB * 8) : (FFN + n0 + (ngB - 8) * 8);

    // A-staging: 2 16B chunks per thread
    const int rA0 = tid >> 2, cA0 = tid & 3;           // idx = tid
    const int rA1 = (tid + 256) >> 2, cA1 = tid & 3;   // idx = tid+256

    v8f accA[4]; v8f accB[4];
#pragma unroll
    for (int i = 0; i < 4; ++i) { accA[i] = (v8f)0.f; accB[i] = (v8f)0.f; }

    const int hlf = lane >> 4;
    const int rr  = lane & 15;
    const int kbA = hlf * 16;
    const int kbB = hlf * 16;

    auto stage = [&](int buf, int k0) {
        // A tile: async 16B copies (bf16 source, no transpose)
        async_copy16((unsigned)(uintptr_t)&ldsA[buf][rA0 * LDST + cA0 * 4],
                     xb + (size_t)(m0 + rA0) * (HIDDEN / 2) + (k0 >> 1) + cA0 * 4);
        async_copy16((unsigned)(uintptr_t)&ldsA[buf][rA1 * LDST + cA1 * 4],
                     xb + (size_t)(m0 + rA1) * (HIDDEN / 2) + (k0 >> 1) + cA1 * 4);
        // B tile transposed: two K-rows x 8 cols -> 8 packed dwords
        uint4 q0 = *(const uint4*)&w1b[(size_t)(k0 + 2 * kpB) * FFN2 + colB];
        uint4 q1 = *(const uint4*)&w1b[(size_t)(k0 + 2 * kpB + 1) * FFN2 + colB];
        const unsigned short* h0 = (const unsigned short*)&q0;
        const unsigned short* h1 = (const unsigned short*)&q1;
#pragma unroll
        for (int j = 0; j < 8; ++j)
            ldsB[buf][(ngB * 8 + j) * LDST + kpB] = (unsigned)h0[j] | ((unsigned)h1[j] << 16);
    };

    stage(0, 0);
    wait_async0();
    __syncthreads();

    int cur = 0;
    for (int k0 = 0; k0 < HIDDEN; k0 += 32) {
        if (k0 + 32 < HIDDEN) stage(cur ^ 1, k0 + 32);   // overlap with compute

        const unsigned* As = &ldsA[cur][0];
        const unsigned* Bs = &ldsB[cur][0];
        FragU afr[4];
#pragma unroll
        for (int mf = 0; mf < 4; ++mf) {
            const unsigned* p = &As[(wm * 64 + mf * 16 + rr) * LDST + (kbA >> 1)];
            afr[mf].q[0] = *(const uint4*)p;
            afr[mf].q[1] = *(const uint4*)(p + 4);
        }
        FragU bfa, bfb;
        {
            const unsigned* p = &Bs[(wn * 16 + rr) * LDST + (kbB >> 1)];
            bfa.q[0] = *(const uint4*)p;  bfa.q[1] = *(const uint4*)(p + 4);
            const unsigned* q = &Bs[(64 + wn * 16 + rr) * LDST + (kbB >> 1)];
            bfb.q[0] = *(const uint4*)q;  bfb.q[1] = *(const uint4*)(q + 4);
        }
#pragma unroll
        for (int mf = 0; mf < 4; ++mf) {
            accA[mf] = __builtin_amdgcn_wmma_f32_16x16x32_bf16(
                false, afr[mf].v, false, bfa.v, (short)0, accA[mf], false, false);
            accB[mf] = __builtin_amdgcn_wmma_f32_16x16x32_bf16(
                false, afr[mf].v, false, bfb.v, (short)0, accB[mf], false, false);
        }
        wait_async0();
        __syncthreads();
        cur ^= 1;
    }

    // epilogue: act = silu(a)*b, bf16
    const int colF = n0 + wn * 16 + (lane & 15);
#pragma unroll
    for (int mf = 0; mf < 4; ++mf) {
#pragma unroll
        for (int i = 0; i < 8; ++i) {
            int r = m0 + wm * 64 + mf * 16 + hlf * 8 + i;
            float a = accA[mf][i], b = accB[mf][i];
            float sa = a * __builtin_amdgcn_rcpf(1.f + __expf(-a));
            act[(size_t)r * FFN + colF] = f2bf(sa * b);
        }
    }
}

// ---------------- GEMM2: out += C[t,e] * (act @ w2[e]) ----------------
__global__ __launch_bounds__(256) void moe_gemm2_kernel(const unsigned* __restrict__ actdw,
                                                        const unsigned short* __restrict__ w2b,
                                                        const float* __restrict__ C,
                                                        float* __restrict__ out,
                                                        int e) {
    __shared__ __attribute__((aligned(16))) unsigned ldsA[2][128 * LDST];
    __shared__ __attribute__((aligned(16))) unsigned ldsB[2][128 * LDST];

    const int tid  = threadIdx.x;
    const int lane = tid & 31;
    const int wave = tid >> 5;
    const int wm   = wave >> 2;
    const int wn   = wave & 3;
    const int m0   = blockIdx.y * 128;
    const int n1   = blockIdx.x * 128;

    const int kpB = tid >> 4;
    const int ngB = tid & 15;
    const int colB = n1 + ngB * 8;

    const int rA0 = tid >> 2, cA0 = tid & 3;
    const int rA1 = (tid + 256) >> 2;

    v8f acc[4][2];
#pragma unroll
    for (int i = 0; i < 4; ++i) { acc[i][0] = (v8f)0.f; acc[i][1] = (v8f)0.f; }

    const int hlf = lane >> 4;
    const int rr  = lane & 15;
    const int kbA = hlf * 16;
    const int kbB = hlf * 16;

    auto stage = [&](int buf, int k0) {
        async_copy16((unsigned)(uintptr_t)&ldsA[buf][rA0 * LDST + cA0 * 4],
                     actdw + (size_t)(m0 + rA0) * (FFN / 2) + (k0 >> 1) + cA0 * 4);
        async_copy16((unsigned)(uintptr_t)&ldsA[buf][rA1 * LDST + cA0 * 4],
                     actdw + (size_t)(m0 + rA1) * (FFN / 2) + (k0 >> 1) + cA0 * 4);
        uint4 q0 = *(const uint4*)&w2b[(size_t)(k0 + 2 * kpB) * HIDDEN + colB];
        uint4 q1 = *(const uint4*)&w2b[(size_t)(k0 + 2 * kpB + 1) * HIDDEN + colB];
        const unsigned short* h0 = (const unsigned short*)&q0;
        const unsigned short* h1 = (const unsigned short*)&q1;
#pragma unroll
        for (int j = 0; j < 8; ++j)
            ldsB[buf][(ngB * 8 + j) * LDST + kpB] = (unsigned)h0[j] | ((unsigned)h1[j] << 16);
    };

    stage(0, 0);
    wait_async0();
    __syncthreads();

    int cur = 0;
    for (int k0 = 0; k0 < FFN; k0 += 32) {
        if (k0 + 32 < FFN) stage(cur ^ 1, k0 + 32);

        const unsigned* As = &ldsA[cur][0];
        const unsigned* Bs = &ldsB[cur][0];
        FragU afr[4];
#pragma unroll
        for (int mf = 0; mf < 4; ++mf) {
            const unsigned* p = &As[(wm * 64 + mf * 16 + rr) * LDST + (kbA >> 1)];
            afr[mf].q[0] = *(const uint4*)p;
            afr[mf].q[1] = *(const uint4*)(p + 4);
        }
        FragU bfr[2];
#pragma unroll
        for (int nf = 0; nf < 2; ++nf) {
            const unsigned* p = &Bs[(wn * 32 + nf * 16 + rr) * LDST + (kbB >> 1)];
            bfr[nf].q[0] = *(const uint4*)p;  bfr[nf].q[1] = *(const uint4*)(p + 4);
        }
#pragma unroll
        for (int mf = 0; mf < 4; ++mf)
#pragma unroll
            for (int nf = 0; nf < 2; ++nf)
                acc[mf][nf] = __builtin_amdgcn_wmma_f32_16x16x32_bf16(
                    false, afr[mf].v, false, bfr[nf].v, (short)0, acc[mf][nf], false, false);

        wait_async0();
        __syncthreads();
        cur ^= 1;
    }

    const int cc = lane & 15;
#pragma unroll
    for (int mf = 0; mf < 4; ++mf) {
#pragma unroll
        for (int i = 0; i < 8; ++i) {
            int t  = m0 + wm * 64 + mf * 16 + hlf * 8 + i;
            float cw = C[(size_t)t * NEXP + e];
            int bb = t >> 11, ss = t & (SEQ - 1);
            size_t orow = ((size_t)ss * BATCH + bb) * HIDDEN;
#pragma unroll
            for (int nf = 0; nf < 2; ++nf) {
                int hc = n1 + wn * 32 + nf * 16 + cc;
                float* po = out + orow + hc;
                *po = *po + cw * acc[mf][nf][i];
            }
        }
    }
}

extern "C" void kernel_launch(void* const* d_in, const int* in_sizes, int n_in,
                              void* d_out, int out_size, void* d_ws, size_t ws_size,
                              hipStream_t stream) {
    const float* x  = (const float*)d_in[0];   // [SEQ, BATCH, HIDDEN]
    const float* wg = (const float*)d_in[1];   // [NEXP, HIDDEN]
    const float* w1 = (const float*)d_in[2];   // [NEXP, HIDDEN, 2*FFN]
    const float* w2 = (const float*)d_in[3];   // [NEXP, FFN, HIDDEN]
    float* out = (float*)d_out;                // [SEQ, BATCH, HIDDEN]

    // workspace layout (all 16B aligned)
    char* ws = (char*)d_ws;
    float*          Cws   = (float*)ws;                                   // 256 KB
    unsigned*       xb    = (unsigned*)(ws + 262144);                     // 32 MB  (bf16 pairs)
    unsigned short* actws = (unsigned short*)(ws + 262144 + 33554432);    // 22 MB
    unsigned*       w1b   = (unsigned*)(ws + 262144 + 33554432 + 23068672);        // 11.5 MB
    unsigned*       w2b   = (unsigned*)(ws + 262144 + 33554432 + 23068672 + 11534336); // 5.8 MB

    moe_zero_kernel<<<16384, 256, 0, stream>>>((float4*)out);
    moe_gate_kernel<<<NTOK / 8, 256, 0, stream>>>(x, wg, Cws);
    moe_convx_kernel<<<NTOK, 256, 0, stream>>>(x, xb);

    for (int e = 0; e < NEXP; ++e) {
        moe_convw_kernel<<<(HIDDEN * FFN2 / 2) / 1024, 256, 0, stream>>>(
            w1 + (size_t)e * HIDDEN * FFN2, w1b);
        moe_convw_kernel<<<(FFN * HIDDEN / 2) / 1024, 256, 0, stream>>>(
            w2 + (size_t)e * FFN * HIDDEN, w2b);
        moe_gemm1_kernel<<<dim3(FFN / 64, NTOK / 128), 256, 0, stream>>>(
            xb, (const unsigned short*)w1b, actws);
        moe_gemm2_kernel<<<dim3(HIDDEN / 128, NTOK / 128), 256, 0, stream>>>(
            (const unsigned*)actws, (const unsigned short*)w2b, Cws, out, e);
    }
}
